// spatial_strip_att_dilated_52905407152335
// MI455X (gfx1250) — matrix operand order repaired and checked by reading, not compiled, not run on hardware
//
#include <hip/hip_runtime.h>
#include <hip/hip_bf16.h>

typedef __attribute__((ext_vector_type(16))) _Float16 v16h;
typedef __attribute__((ext_vector_type(8)))  float    v8f;
typedef __attribute__((ext_vector_type(4)))  float    f4;

#define NB   16     // batches
#define NC   256    // channels
#define HW   128    // height and width
#define HWSQ (HW * HW)
#define KTAP 5
#define GRP  2
#define ROWS_PER_BLK 8
#define CONV_BLOCKS (NB * NC * (HW / ROWS_PER_BLK))   // 65536

// ---------------------------------------------------------------------------
// Pass 1: per-(n,c) spatial mean.  One block per (n,c); 256 threads each read
// 16 coalesced float4 (64 KiB/block), LDS tree-reduce, write mean.
// Default (RT) temporal hint: we WANT the tail of x resident in the 192 MB L2
// for pass 3's reverse-order sweep.
// ---------------------------------------------------------------------------
__global__ void ssad_avg_kernel(const float* __restrict__ x,
                                float* __restrict__ avg) {
    __shared__ float red[256];
    const int nc = blockIdx.x;                 // 0 .. 4095
    const int t  = threadIdx.x;
    const f4* p = (const f4*)(x + (size_t)nc * HWSQ);
    float s = 0.f;
#pragma unroll
    for (int i = 0; i < 16; ++i) {
        f4 v = p[i * 256 + t];
        s += v.x + v.y + v.z + v.w;
    }
    red[t] = s;
    __syncthreads();
    for (int off = 128; off > 0; off >>= 1) {
        if (t < off) red[t] += red[t + off];
        __syncthreads();
    }
    if (t == 0) avg[nc] = red[0] * (1.0f / (float)HWSQ);
}

// ---------------------------------------------------------------------------
// Pass 2: single-wave WMMA GEMM  f = sigmoid(avg(16x256) * W^T(256x16pad)).
// V_WMMA_F32_16X16X32_F16, K=256 in 8 chunks of 32.
// A (16-bit 16x32): lane<16 -> row M=lane, elems 0..7 -> K=base+hi*8+0..7,
//   elems 8..15 -> K=base+hi*8+16..23 (hi lanes shift K ranges by 8).
// B (16-bit 32x16): lane -> col N=lane&15, elems -> same K mapping as A.
// All loads are unconditional float4 from clamped (always-valid) addresses;
// padding columns are zeroed with a multiply -> no EXEC-mask branches.
// ---------------------------------------------------------------------------
__device__ __forceinline__ void cvt8(v16h& d, int off, f4 p, f4 q) {
    d[off + 0] = (_Float16)p.x; d[off + 1] = (_Float16)p.y;
    d[off + 2] = (_Float16)p.z; d[off + 3] = (_Float16)p.w;
    d[off + 4] = (_Float16)q.x; d[off + 5] = (_Float16)q.y;
    d[off + 6] = (_Float16)q.z; d[off + 7] = (_Float16)q.w;
}

__global__ void ssad_filt_wmma_kernel(const float* __restrict__ avg,
                                      const float* __restrict__ W,
                                      float* __restrict__ fcoef) {
    const int lane = threadIdx.x;      // 0..31, full wave (EXEC all ones)
    const int row  = lane & 15;        // A: batch row / B: output column
    const int hi   = lane >> 4;

    const float  bmask = (row < GRP * KTAP) ? 1.f : 0.f;
    const float* wrow  = W + (row < GRP * KTAP ? row : 0) * NC; // clamped
    const float* arow  = avg + row * NC;

    v8f acc = {};
#pragma unroll
    for (int kc = 0; kc < 8; ++kc) {
        const int base = kc * 32;
        // A: two contiguous 8-float runs per lane
        f4 a0 = *(const f4*)(arow + base + hi * 8 + 0);
        f4 a1 = *(const f4*)(arow + base + hi * 8 + 4);
        f4 a2 = *(const f4*)(arow + base + hi * 8 + 16);
        f4 a3 = *(const f4*)(arow + base + hi * 8 + 20);
        // B: two contiguous 8-float runs per lane (same K as A), zero-masked for pad cols
        f4 b0 = *(const f4*)(wrow + base + hi * 8 + 0)  * bmask;
        f4 b1 = *(const f4*)(wrow + base + hi * 8 + 4)  * bmask;
        f4 b2 = *(const f4*)(wrow + base + hi * 8 + 16) * bmask;
        f4 b3 = *(const f4*)(wrow + base + hi * 8 + 20) * bmask;

        v16h a, b;
        cvt8(a, 0, a0, a1);
        cvt8(a, 8, a2, a3);
        cvt8(b, 0, b0, b1);
        cvt8(b, 8, b2, b3);

        acc = __builtin_amdgcn_wmma_f32_16x16x32_f16(
            /*neg_a=*/false, a, /*neg_b=*/false, b,
            /*c_mod=*/(short)0, acc, /*reuse_a=*/false, /*reuse_b=*/false);
    }

#pragma unroll
    for (int v = 0; v < 8; ++v) {
        const int n = v + 8 * hi;                  // batch index (M)
        const int o = row;                         // output index (N)
        const float sg = 1.0f / (1.0f + __expf(-acc[v]));
        fcoef[n * 16 + o] = sg;                    // 16x16, first 10 cols live
    }
}

// ---------------------------------------------------------------------------
// Pass 3: dilated (d=2) 5-tap reflect-padded conv along width.
// One block per (n, c, 8-row tile): stage 8x128 f32 tile in LDS (4 KiB),
// 4 outputs/thread, float4 NT load + float4 NT store.
//
// Block order is REVERSED: pass 1 streamed x ascending, so the tail ~192 MB
// of x is still resident in L2. First-dispatched conv blocks consume that
// tail -> large fraction of the re-read comes from L2 instead of HBM.
// NT stores keep `out` (write-once, never read) from evicting x lines.
// ---------------------------------------------------------------------------
__device__ __forceinline__ int reflect_idx(int i) {
    i = i < 0 ? -i : i;
    return i > (HW - 1) ? (2 * (HW - 1) - i) : i;
}

__global__ void ssad_conv_kernel(const float* __restrict__ x,
                                 const float* __restrict__ fcoef,
                                 float* __restrict__ out) {
    __shared__ float tile[ROWS_PER_BLK * HW];   // 4 KiB
    const int bi = (CONV_BLOCKS - 1) - (int)blockIdx.x;   // reverse sweep
    const int n  = bi >> 12;                    // / (256*16)
    const int c  = (bi >> 4) & 255;
    const int ht = bi & 15;
    const int t  = threadIdx.x;

    const size_t base = ((size_t)(n * NC + c) * HW + ht * ROWS_PER_BLK) * HW;

    // stage tile; x is last-use here -> nontemporal load
    f4 ld = __builtin_nontemporal_load((const f4*)(x + base) + t);
    ((f4*)tile)[t] = ld;

    // block-uniform filter taps (channel parity selects group)
    const float* fc = fcoef + n * 16 + (c & 1) * KTAP;
    const float f0 = fc[0], f1 = fc[1], f2 = fc[2], f3 = fc[3], f4_ = fc[4];

    __syncthreads();

    const int row = t >> 5;                     // 0..7 within tile
    const int w0  = (t & 31) * 4;
    const float* s = tile + row * HW;

    f4 o;
#pragma unroll
    for (int i = 0; i < 4; ++i) {
        const int w = w0 + i;
        float acc;
        acc  = f0  * s[reflect_idx(w - 4)];
        acc += f1  * s[reflect_idx(w - 2)];
        acc += f2  * s[w];
        acc += f3  * s[reflect_idx(w + 2)];
        acc += f4_ * s[reflect_idx(w + 4)];
        o[i] = acc;
    }
    __builtin_nontemporal_store(o, (f4*)(out + base) + t);
}

// ---------------------------------------------------------------------------
extern "C" void kernel_launch(void* const* d_in, const int* in_sizes, int n_in,
                              void* d_out, int out_size, void* d_ws, size_t ws_size,
                              hipStream_t stream) {
    const float* x = (const float*)d_in[0];   // (16,256,128,128) f32
    const float* W = (const float*)d_in[1];   // (10,256) f32
    float* out = (float*)d_out;

    float* avg   = (float*)d_ws;              // 4096 floats
    float* fcoef = avg + NB * NC;             // 256 floats (16x16)

    // Pass 1: means (ascending stream; leaves tail of x in L2)
    ssad_avg_kernel<<<NB * NC, 256, 0, stream>>>(x, avg);

    // Pass 2: WMMA filter GEMM + sigmoid (one wave)
    ssad_filt_wmma_kernel<<<1, 32, 0, stream>>>(avg, W, fcoef);

    // Pass 3: dilated conv, reverse-order sweep for L2 tail reuse
    ssad_conv_kernel<<<CONV_BLOCKS, 256, 0, stream>>>(x, fcoef, out);
}
